// ResMamba_63720134803771
// MI455X (gfx1250) — matrix operand order repaired and checked by reading, not compiled, not run on hardware
//
#include <hip/hip_runtime.h>
#include <hip/hip_bf16.h>
#include <math.h>

// ---------------- types ----------------
typedef __bf16 bf16_t;
typedef bf16_t v16bf __attribute__((ext_vector_type(16)));
typedef bf16_t v8bf  __attribute__((ext_vector_type(8)));
typedef float  v8f   __attribute__((ext_vector_type(8)));
typedef float  f32x4 __attribute__((ext_vector_type(4)));

#define BATCH   2
#define SEQLEN  2048
#define MTOK    4096            // BATCH * SEQLEN; always a multiple of BM
#define EPSF    1e-5f
#define SELU_SCALE 1.0507009873554805f
#define SELU_ALPHA 1.6732632423543772f

#define BM 128
#define BN 128
#define BK 32

// ==========================================================================
// GEMM: C[M,N] = A[M,K] * W[N,K]^T   (A,W f32 row-major; bf16 WMMA, f32 acc)
// M is guaranteed a multiple of BM (4096 tokens) -> no M guards.
// N guarded branch-free: clamped row + select-zero.
// Double-buffered LDS tiles, one barrier per K step.
// mode 0: plain store; mode 1: fused batchnorm + SELU epilogue (per-col)
// ==========================================================================
__global__ __launch_bounds__(256) void gemm_bf16_wmma_k(
    const float* __restrict__ A, const float* __restrict__ W, float* __restrict__ C,
    int N, int K, int lda, int ldc,
    const float* __restrict__ gamma, const float* __restrict__ beta,
    const float* __restrict__ mean, const float* __restrict__ var, int mode)
{
    __shared__ bf16_t sA[2][BM][BK + 8];   // +8 bf16 pad: 80B row stride
    __shared__ bf16_t sW[2][BN][BK + 8];

    const int t    = threadIdx.x;
    const int lane = t & 31;
    const int wv   = t >> 5;            // 8 waves
    const int wm   = wv & 3;            // 4 waves along M (32 rows each)
    const int wn   = wv >> 2;           // 2 waves along N (64 cols each)
    const int bMr  = blockIdx.y * BM;
    const int bNr  = blockIdx.x * BN;
    const int mr   = lane & 15;
    const int half = lane >> 4;

    // ---- per-thread static load coordinates (4 float4 per tile each) ----
    int lrow[4], lc4[4], wok[4];
    const float* aptr[4];
    const float* wptr[4];
#pragma unroll
    for (int i = 0; i < 4; ++i) {
        int idx = t + (i << 8);
        int row = idx >> 3;             // 0..127
        int c4  = (idx & 7) << 2;       // 0,4,...,28
        lrow[i] = row; lc4[i] = c4;
        aptr[i] = A + (size_t)(bMr + row) * lda + c4;
        int gn  = bNr + row;
        wok[i]  = (gn < N);
        if (gn >= N) gn = N - 1;        // clamp: load stays in-bounds
        wptr[i] = W + (size_t)gn * K + c4;
    }
    // prefetch pointer: waves 0-3 prefetch A rows, waves 4-7 prefetch W rows
    int prow = t & 127;
    int pwn  = bNr + prow; if (pwn >= N) pwn = N - 1;
    const float* pf = (wv < 4) ? (A + (size_t)(bMr + prow) * lda)
                               : (W + (size_t)pwn * K);

    auto loadrv = [&](int k0, f32x4* ra, f32x4* rw) {
#pragma unroll
        for (int i = 0; i < 4; ++i) {
            ra[i] = *(const f32x4*)(aptr[i] + k0);
            f32x4 v = *(const f32x4*)(wptr[i] + k0);
            if (!wok[i]) { v[0] = 0.f; v[1] = 0.f; v[2] = 0.f; v[3] = 0.f; }
            rw[i] = v;
        }
    };
    auto storelds = [&](int buf, const f32x4* ra, const f32x4* rw) {
#pragma unroll
        for (int i = 0; i < 4; ++i) {
            int row = lrow[i], c4 = lc4[i];
            sA[buf][row][c4 + 0] = (bf16_t)ra[i][0];
            sA[buf][row][c4 + 1] = (bf16_t)ra[i][1];
            sA[buf][row][c4 + 2] = (bf16_t)ra[i][2];
            sA[buf][row][c4 + 3] = (bf16_t)ra[i][3];
            sW[buf][row][c4 + 0] = (bf16_t)rw[i][0];
            sW[buf][row][c4 + 1] = (bf16_t)rw[i][1];
            sW[buf][row][c4 + 2] = (bf16_t)rw[i][2];
            sW[buf][row][c4 + 3] = (bf16_t)rw[i][3];
        }
    };

    v8f acc[2][4];
#pragma unroll
    for (int i = 0; i < 2; ++i)
#pragma unroll
        for (int j = 0; j < 4; ++j)
#pragma unroll
            for (int r = 0; r < 8; ++r) acc[i][j][r] = 0.0f;

    const int nk = K / BK;              // K always a multiple of 32
    f32x4 rA[4], rW[4];
    loadrv(0, rA, rW);
    storelds(0, rA, rW);
    __syncthreads();

    for (int kt = 0; kt < nk; ++kt) {
        const int cur = kt & 1;
        const bool hasNext = (kt + 1) < nk;
        if (hasNext) {
            loadrv((kt + 1) * BK, rA, rW);          // overlap with WMMAs below
            if (kt + 2 < nk)
                __builtin_prefetch(pf + (size_t)(kt + 2) * BK, 0, 1);
        }

        // A fragment (16x32): lane<16 -> K[0..7],K[16..23]; lane>=16 -> K[8..15],K[24..31]
        v16bf af[2], bfr[4];
#pragma unroll
        for (int i = 0; i < 2; ++i) {
            int row = wm * 32 + i * 16 + mr;
            v8bf lo = *(const v8bf*)&sA[cur][row][half * 8];
            v8bf hi = *(const v8bf*)&sA[cur][row][16 + half * 8];
            af[i] = __builtin_shufflevector(lo, hi, 0,1,2,3,4,5,6,7,8,9,10,11,12,13,14,15);
        }
        // B fragment (32x16): lane<16 -> K[0..7],K[16..23] of col n; lane>=16 -> K[8..15],K[24..31]
#pragma unroll
        for (int j = 0; j < 4; ++j) {
            int row = wn * 64 + j * 16 + mr;
            v8bf lo = *(const v8bf*)&sW[cur][row][half * 8];
            v8bf hi = *(const v8bf*)&sW[cur][row][16 + half * 8];
            bfr[j] = __builtin_shufflevector(lo, hi, 0,1,2,3,4,5,6,7,8,9,10,11,12,13,14,15);
        }
#pragma unroll
        for (int i = 0; i < 2; ++i)
#pragma unroll
            for (int j = 0; j < 4; ++j)
                acc[i][j] = __builtin_amdgcn_wmma_f32_16x16x32_bf16(
                    false, af[i], false, bfr[j], (short)0, acc[i][j], false, false);

        if (hasNext) {
            storelds(1 - cur, rA, rW);              // other buffer: no alias with reads
            __syncthreads();                        // single barrier per K step
        }
    }

    // epilogue: C layout per lane — col = lane&15, row = r + 8*(lane>>4)
#pragma unroll
    for (int j = 0; j < 4; ++j) {
        int col = bNr + wn * 64 + j * 16 + mr;
        if (col >= N) continue;
        float sc = 1.f, sh = 0.f;
        if (mode == 1) {
            float s = rsqrtf(var[col] + EPSF) * gamma[col];
            sc = s;
            sh = beta[col] - mean[col] * s;
        }
#pragma unroll
        for (int i = 0; i < 2; ++i) {
            int rowb = bMr + wm * 32 + i * 16 + (half << 3);
#pragma unroll
            for (int r = 0; r < 8; ++r) {
                float v = acc[i][j][r];
                if (mode == 1) {
                    v = v * sc + sh;
                    v = (v > 0.f) ? SELU_SCALE * v
                                  : SELU_SCALE * SELU_ALPHA * (__expf(v) - 1.f);
                }
                C[(size_t)(rowb + r) * ldc + col] = v;
            }
        }
    }
}

// ==========================================================================
// Depthwise causal conv (k=4) + bias + SiLU over the xBC slice of zxbcdt
// ==========================================================================
__global__ void conv_silu_k(const float* __restrict__ zx, const float* __restrict__ w,
                            const float* __restrict__ bias, float* __restrict__ out,
                            int L, int conv_dim, int d_in_proj, int d_inner, int total)
{
    int idx = blockIdx.x * blockDim.x + threadIdx.x;
    if (idx >= total) return;
    int c = idx % conv_dim;
    int l = (idx / conv_dim) % L;
    int b = idx / (conv_dim * L);
    const float* base = zx + ((size_t)b * L) * d_in_proj + d_inner + c;
    float acc = bias[c];
#pragma unroll
    for (int i = 0; i < 4; ++i) {
        int ls = l - 3 + i;
        if (ls >= 0) acc = fmaf(base[(size_t)ls * d_in_proj], w[c * 4 + i], acc);
    }
    out[idx] = acc / (1.f + __expf(-acc));   // silu
}

// ==========================================================================
// Selective scan: one (batch, head) per 64-thread block; thread p owns state
// row h[p][0..63] in registers; B_t/C_t broadcast via LDS.
// ==========================================================================
__global__ void ssm_scan_k(const float* __restrict__ zx, const float* __restrict__ xc,
                           const float* __restrict__ dt_bias, const float* __restrict__ A_log,
                           const float* __restrict__ Dp, float* __restrict__ y,
                           int L, int nh, int d_in_proj, int conv_dim)
{
    const int b = blockIdx.x / nh;
    const int h = blockIdx.x % nh;
    const int p = threadIdx.x;
    const int d_inner = nh << 6;

    const float Ah  = -__expf(A_log[h]);
    const float dtb = dt_bias[h];
    const float Dh  = Dp[h];

    float hs[64];
#pragma unroll
    for (int n = 0; n < 64; ++n) hs[n] = 0.f;

    __shared__ float sB[64], sC[64];
    const size_t rb = (size_t)b * L;
    for (int l = 0; l < L; ++l) {
        const size_t r = rb + l;
        float dtr = zx[r * d_in_proj + (d_in_proj - nh) + h] + dtb;
        float dt  = (dtr > 20.f) ? dtr : log1pf(__expf(dtr));   // softplus
        float dA  = __expf(dt * Ah);
        const float* xrow = xc + r * conv_dim;
        float xp = xrow[h * 64 + p];
        sB[p] = xrow[d_inner + p];
        sC[p] = xrow[d_inner + 64 + p];
        __syncthreads();
        float coef = dt * xp;
        float accy = 0.f;
#pragma unroll
        for (int n = 0; n < 64; ++n) {
            hs[n] = fmaf(hs[n], dA, coef * sB[n]);
            accy  = fmaf(hs[n], sC[n], accy);
        }
        y[r * d_inner + h * 64 + p] = accy + xp * Dh;
        __syncthreads();
    }
}

// ==========================================================================
// y *= silu(z); RMSNorm over d_inner; scale by norm_w.  One token row/block.
// ==========================================================================
__global__ void gate_rmsnorm_k(const float* __restrict__ ybuf, const float* __restrict__ zx,
                               const float* __restrict__ norm_w, float* __restrict__ out,
                               int d_inner, int d_in_proj)
{
    const int row = blockIdx.x;
    const float* yr = ybuf + (size_t)row * d_inner;
    const float* zr = zx + (size_t)row * d_in_proj;
    const int per = d_inner >> 8;          // d_inner is 512..4096, /256
    float g[16];
    float ss = 0.f;
    for (int i = 0; i < per; ++i) {
        int c = threadIdx.x + (i << 8);
        float z = zr[c];
        float v = yr[c] * (z / (1.f + __expf(-z)));
        g[i] = v;
        ss = fmaf(v, v, ss);
    }
    __shared__ float red[256];
    red[threadIdx.x] = ss;
    __syncthreads();
    for (int s = 128; s > 0; s >>= 1) {
        if (threadIdx.x < s) red[threadIdx.x] += red[threadIdx.x + s];
        __syncthreads();
    }
    float rms = rsqrtf(red[0] / (float)d_inner + EPSF);
    for (int i = 0; i < per; ++i) {
        int c = threadIdx.x + (i << 8);
        out[(size_t)row * d_inner + c] = g[i] * rms * norm_w[c];
    }
}

__global__ void add_k(const float* __restrict__ a, const float* __restrict__ b,
                      float* __restrict__ o, int n)
{
    int i = blockIdx.x * blockDim.x + threadIdx.x;
    if (i < n) o[i] = a[i] + b[i];
}

// ==========================================================================
// Host orchestration
// ==========================================================================
struct MambaP { const float *in_proj, *conv_w, *conv_b, *dt_bias, *A_log, *D, *norm_w, *out_proj; };
struct ProjP  { const float *w, *gamma, *beta, *mean, *var; };

extern "C" void kernel_launch(void* const* d_in, const int* in_sizes, int n_in,
                              void* d_out, int out_size, void* d_ws, size_t ws_size,
                              hipStream_t stream)
{
    (void)in_sizes; (void)n_in; (void)out_size; (void)ws_size;
    const float* x0 = (const float*)d_in[0];

    int idx = 1;
    auto nextp = [&]() { return (const float*)d_in[idx++]; };
    auto takeM = [&](MambaP& m) {
        m.in_proj = nextp(); m.conv_w = nextp(); m.conv_b = nextp(); m.dt_bias = nextp();
        m.A_log = nextp(); m.D = nextp(); m.norm_w = nextp(); m.out_proj = nextp();
    };
    auto takeP = [&](ProjP& p) {
        p.w = nextp(); p.gamma = nextp(); p.beta = nextp(); p.mean = nextp(); p.var = nextp();
    };
    MambaP um[4], dm[4]; ProjP up[4], dp[4];
    for (int i = 0; i < 4; ++i) { takeM(um[i]); takeP(up[i]); }   // up: mamba, proj
    for (int i = 0; i < 4; ++i) { takeP(dp[i]); takeM(dm[i]); }   // down: proj, mamba

    const size_t M = MTOK;
    size_t off = 0;
    auto wsf = [&](size_t elems) {
        float* p = (float*)((char*)d_ws + off);
        off += (elems * sizeof(float) + 255) & ~(size_t)255;
        return p;
    };
    float* zx    = wsf(M * 8384);   // max d_in_proj (d=2048)
    float* xconv = wsf(M * 4224);   // max conv_dim
    float* ybuf  = wsf(M * 4096);   // max d_inner
    float* gbuf  = wsf(M * 4096);
    float* tmpA  = wsf(M * 4096);   // mamba out / residual-add buffer
    float* tmpB  = wsf(M * 2048);   // down-path proj out
    float* res[4];
    res[0] = wsf(M * 512); res[1] = wsf(M * 1024);
    res[2] = wsf(M * 2048); res[3] = wsf(M * 4096);

    auto gemm = [&](const float* A, const float* W, float* C, int m, int n, int k,
                    int lda, int ldc, const ProjP* bn) {
        dim3 grid((n + BN - 1) / BN, (m + BM - 1) / BM);
        gemm_bf16_wmma_k<<<grid, 256, 0, stream>>>(A, W, C, n, k, lda, ldc,
            bn ? bn->gamma : nullptr, bn ? bn->beta : nullptr,
            bn ? bn->mean : nullptr, bn ? bn->var : nullptr, bn ? 1 : 0);
    };

    auto mamba = [&](const float* xin, int d, const MambaP& m, float* out) {
        const int d_inner = 2 * d, nh = d / 32;
        const int dip = 4 * d + 128 + nh, cdim = 2 * d + 128;
        gemm(xin, m.in_proj, zx, (int)M, dip, d, d, dip, nullptr);
        int total = (int)(M * (size_t)cdim);
        conv_silu_k<<<(total + 255) / 256, 256, 0, stream>>>(
            zx, m.conv_w, m.conv_b, xconv, SEQLEN, cdim, dip, d_inner, total);
        ssm_scan_k<<<BATCH * nh, 64, 0, stream>>>(
            zx, xconv, m.dt_bias, m.A_log, m.D, ybuf, SEQLEN, nh, dip, cdim);
        gate_rmsnorm_k<<<(int)M, 256, 0, stream>>>(ybuf, zx, m.norm_w, gbuf, d_inner, dip);
        gemm(gbuf, m.out_proj, out, (int)M, d, d_inner, d_inner, d, nullptr);
    };

    // ---------------- up path ----------------
    const float* cur = x0;
    int d = 256;
    for (int i = 0; i < 4; ++i) {
        mamba(cur, d, um[i], tmpA);
        gemm(tmpA, up[i].w, res[i], (int)M, 2 * d, d, d, 2 * d, &up[i]);
        cur = res[i];
        d *= 2;
    }
    // ---------------- down path ----------------
    for (int i = 0; i < 4; ++i) {
        int n = (int)(M * (size_t)d);
        add_k<<<(n + 255) / 256, 256, 0, stream>>>(cur, res[3 - i], tmpA, n);
        gemm(tmpA, dp[i].w, tmpB, (int)M, d / 2, d, d, d / 2, &dp[i]);
        d /= 2;
        float* mout = (i == 3) ? (float*)d_out : tmpA;
        mamba(tmpB, d, dm[i], mout);
        cur = mout;
    }
}
